// TinyLLMBlock_45913200395019
// MI455X (gfx1250) — matrix operand order repaired and checked by reading, not compiled, not run on hardware
//
#include <hip/hip_runtime.h>
#include <hip/hip_bf16.h>
#include <math.h>

// Problem sizes (fixed by the reference)
#define H_    1024
#define NH_   16
#define HD_   64
#define FF_   4096
#define SEQ_  2048
#define BSZ_  2
#define TOK_  (BSZ_*SEQ_)
#define LDQKV 3072            // fused q|k|v row stride

typedef unsigned short ushort_t;
typedef __attribute__((ext_vector_type(16))) __bf16 v16bf;
typedef __attribute__((ext_vector_type(8)))  float  v8f;

union FragBF {            // one WMMA 16x16x32 bf16 operand (8 VGPRs / lane)
  v16bf   v;
  uint4   q[2];
  ushort_t s[16];
};

__device__ __forceinline__ ushort_t f2bf(float f) {
  union { float f; unsigned u; } c; c.f = f;
  unsigned u = c.u;
  u += 0x7fffu + ((u >> 16) & 1u);      // round-to-nearest-even
  return (ushort_t)(u >> 16);
}

// ---- DPP16 butterfly reduction within each 16-lane group (no LDS traffic) ----
// steps: quad_perm xor1 (0xB1), quad_perm xor2 (0x4E), row_half_mirror (0x141),
// row_mirror (0x140). Each step pairs a lane with one in the other half of its
// subgroup, so max/sum butterflies leave every lane with the group-of-16 result.
template<int CTRL>
__device__ __forceinline__ float dppf(float x) {
  return __int_as_float(
      __builtin_amdgcn_mov_dpp(__float_as_int(x), CTRL, 0xf, 0xf, true));
}
__device__ __forceinline__ float grp16_max(float v) {
  v = fmaxf(v, dppf<0xB1>(v));
  v = fmaxf(v, dppf<0x4E>(v));
  v = fmaxf(v, dppf<0x141>(v));
  v = fmaxf(v, dppf<0x140>(v));
  return v;
}
__device__ __forceinline__ float grp16_sum(float v) {
  v += dppf<0xB1>(v);
  v += dppf<0x4E>(v);
  v += dppf<0x141>(v);
  v += dppf<0x140>(v);
  return v;
}

// ---------------------------------------------------------------- cast fp32 -> bf16
__global__ __launch_bounds__(256) void cast_bf16_kernel(const float* __restrict__ in,
                                                        ushort_t* __restrict__ out, int n) {
  int i = blockIdx.x * 256 + threadIdx.x;
  if (i < n) out[i] = f2bf(in[i]);
}

// ---------------------------------------------------------------- LayerNorm (fp32 in, bf16 out)
__global__ __launch_bounds__(256) void ln_bf16_kernel(const float* __restrict__ x,
                                                      const float* __restrict__ g,
                                                      const float* __restrict__ bb,
                                                      ushort_t* __restrict__ out) {
  const int t   = blockIdx.x;
  const int tid = threadIdx.x;
  const float4 v = ((const float4*)(x + (size_t)t * H_))[tid];
  float s = v.x + v.y + v.z + v.w;
  float q = v.x * v.x + v.y * v.y + v.z * v.z + v.w * v.w;
  for (int m = 16; m; m >>= 1) { s += __shfl_xor(s, m, 32); q += __shfl_xor(q, m, 32); }
  __shared__ float red_s[8], red_q[8];
  if ((tid & 31) == 0) { red_s[tid >> 5] = s; red_q[tid >> 5] = q; }
  __syncthreads();
  s = 0.f; q = 0.f;
#pragma unroll
  for (int i = 0; i < 8; ++i) { s += red_s[i]; q += red_q[i]; }
  const float mu   = s * (1.0f / H_);
  const float var  = q * (1.0f / H_) - mu * mu;
  const float rstd = rsqrtf(var + 1e-5f);
#pragma unroll
  for (int i = 0; i < 4; ++i) {
    int idx = tid * 4 + i;
    float xv = (&v.x)[i];
    out[(size_t)t * H_ + idx] = f2bf((xv - mu) * rstd * g[idx] + bb[idx]);
  }
}

// ---------------------------------------------------------------- generic bf16 WMMA GEMM
// 128x128 block tile, 8 waves (4M x 2N), wave tile 32x64 (2x4 WMMA frags),
// BK=32, double-buffered LDS with register staging (1 barrier / K-step).
// TRANS_B: B stored [N,K] row-major (compute A @ B^T); else B is [K,N].
template<bool TRANS_B, bool BIAS, bool GELU_, bool RES, bool OUTF, bool OUTB>
__global__ __launch_bounds__(256)
void gemm_bf16_kernel(const ushort_t* __restrict__ A, const ushort_t* __restrict__ Bw,
                      const float* __restrict__ bias, const float* __restrict__ res,
                      float* __restrict__ outF, ushort_t* __restrict__ outB,
                      int M, int N, int K) {
  __shared__ __align__(16) ushort_t As[2][128][40];   // [buf][m][k]
  __shared__ __align__(16) ushort_t Bt[2][128][40];   // [buf][n][k]

  const int tid  = threadIdx.x;
  const int lane = tid & 31;
  const int w    = tid >> 5;
  const int half = lane >> 4;
  const int l16  = lane & 15;
  const int wm   = w >> 1;                 // 0..3
  const int wn   = w & 1;                  // 0..1
  const int m0   = blockIdx.y * 128;
  const int n0   = blockIdx.x * 128;

  // staging coordinates (per thread: 16 contiguous bf16 for A and for B)
  const int arow  = tid >> 1;              // 0..127
  const int apart = (tid & 1) * 16;        // 0 / 16
  const int bk    = tid >> 3;              // 0..31   (!TRANS_B)
  const int bnp   = (tid & 7) * 16;        // 0..112  (!TRANS_B)

  const v8f vz = {0, 0, 0, 0, 0, 0, 0, 0};
  v8f acc[2][4];
#pragma unroll
  for (int i = 0; i < 2; ++i)
#pragma unroll
    for (int j = 0; j < 4; ++j) acc[i][j] = vz;

  uint4 ra[2], rb[2];
  auto g_load = [&](int k0) {
    const uint4* sa = (const uint4*)(A + (size_t)(m0 + arow) * K + k0 + apart);
    ra[0] = sa[0]; ra[1] = sa[1];
    if (TRANS_B) {
      const uint4* sb = (const uint4*)(Bw + (size_t)(n0 + arow) * K + k0 + apart);
      rb[0] = sb[0]; rb[1] = sb[1];
    } else {
      const uint4* sb = (const uint4*)(Bw + (size_t)(k0 + bk) * N + n0 + bnp);
      rb[0] = sb[0]; rb[1] = sb[1];
    }
  };
  auto s_store = [&](int buf) {
    *(uint4*)&As[buf][arow][apart]     = ra[0];
    *(uint4*)&As[buf][arow][apart + 8] = ra[1];
    if (TRANS_B) {
      *(uint4*)&Bt[buf][arow][apart]     = rb[0];
      *(uint4*)&Bt[buf][arow][apart + 8] = rb[1];
    } else {
      union { uint4 u[2]; ushort_t s[16]; } t;
      t.u[0] = rb[0]; t.u[1] = rb[1];
#pragma unroll
      for (int i = 0; i < 16; ++i) Bt[buf][bnp + i][bk] = t.s[i];   // transpose scatter
    }
  };

  g_load(0);
  s_store(0);
  __syncthreads();

  int buf = 0;
  for (int k0 = 0; k0 < K; k0 += 32) {
    const bool more = (k0 + 32) < K;
    if (more) g_load(k0 + 32);             // overlap next tile with WMMAs

    FragBF a[2], b[4];
#pragma unroll
    for (int mt = 0; mt < 2; ++mt) {       // A frag: j0..7 -> k=half*8+j ; j8..15 -> k=16+half*8+j'
      int row = wm * 32 + mt * 16 + l16;
      a[mt].q[0] = *(const uint4*)&As[buf][row][half * 8];
      a[mt].q[1] = *(const uint4*)&As[buf][row][16 + half * 8];
    }
#pragma unroll
    for (int nt = 0; nt < 4; ++nt) {       // B frag: j0..7 -> k=half*8+j ; j8..15 -> k=16+half*8+j'
      int col = wn * 64 + nt * 16 + l16;
      b[nt].q[0] = *(const uint4*)&Bt[buf][col][half * 8];
      b[nt].q[1] = *(const uint4*)&Bt[buf][col][16 + half * 8];
    }
#pragma unroll
    for (int mt = 0; mt < 2; ++mt)
#pragma unroll
      for (int nt = 0; nt < 4; ++nt)
        acc[mt][nt] = __builtin_amdgcn_wmma_f32_16x16x32_bf16(
            false, a[mt].v, false, b[nt].v, (short)0, acc[mt][nt], false, false);

    if (more) s_store(buf ^ 1);            // fill the other buffer
    __syncthreads();
    buf ^= 1;
  }

  // epilogue: C layout -> row = r + half*8, col = lane%16
#pragma unroll
  for (int mt = 0; mt < 2; ++mt)
#pragma unroll
    for (int nt = 0; nt < 4; ++nt) {
      int col = n0 + wn * 64 + nt * 16 + l16;
#pragma unroll
      for (int r = 0; r < 8; ++r) {
        int row = m0 + wm * 32 + mt * 16 + r + half * 8;
        float v = acc[mt][nt][r];
        if (BIAS)  v += bias[col];
        if (GELU_) v = 0.5f * v * (1.0f + erff(v * 0.70710678118654752f));
        if (RES)   v += res[(size_t)row * N + col];
        if (OUTF)  outF[(size_t)row * N + col] = v;
        if (OUTB)  outB[(size_t)row * N + col] = f2bf(v);
      }
    }
}

// ---------------------------------------------------------------- flash attention (bf16 WMMA)
// grid: (SEQ/128, NH, BSZ); each wave owns a 16-row Q tile; block streams 32-key K/V tiles.
// q/k/v live in one fused buffer with row stride ldq.
__global__ __launch_bounds__(256)
void attn_kernel(const ushort_t* __restrict__ qb, const ushort_t* __restrict__ kb,
                 const ushort_t* __restrict__ vb, ushort_t* __restrict__ ob, int ldq) {
  __shared__ __align__(16) ushort_t Kt[32][72];      // [key][d]
  __shared__ __align__(16) ushort_t Vt[64][40];      // [d][key] transposed
  __shared__ __align__(16) ushort_t Pt[8][16][32];   // per-wave P tile [row][key]

  const int tid  = threadIdx.x;
  const int lane = tid & 31;
  const int w    = tid >> 5;
  const int half = lane >> 4;
  const int l16  = lane & 15;
  const int h    = blockIdx.y;
  const int bidx = blockIdx.z;
  const int q0   = blockIdx.x * 128 + w * 16;
  const size_t tokbase = (size_t)bidx * SEQ_;
  const float L2E = 1.4426950408889634f;

  // Q fragments: 2 K-chunks over head-dim 64, kept in registers for the whole pass
  FragBF fq[2];
  {
    const ushort_t* qrow = qb + (tokbase + q0 + l16) * ldq + h * HD_;
#pragma unroll
    for (int c = 0; c < 2; ++c) {
      fq[c].q[0] = *(const uint4*)(qrow + c * 32 + half * 8);
      fq[c].q[1] = *(const uint4*)(qrow + c * 32 + 16 + half * 8);
    }
  }

  const v8f vz = {0, 0, 0, 0, 0, 0, 0, 0};
  float mst[8], lst[8];
  v8f o[4];
#pragma unroll
  for (int i = 0; i < 8; ++i) { mst[i] = -3.0e38f; lst[i] = 0.0f; }
#pragma unroll
  for (int i = 0; i < 4; ++i) o[i] = vz;

  for (int kt = 0; kt < SEQ_ / 32; ++kt) {
    const int key0 = kt * 32;
    { // cooperative stage: K row-major, V transposed
      int row  = tid >> 3;
      int part = (tid & 7) * 8;
      const ushort_t* kr = kb + (tokbase + key0 + row) * ldq + h * HD_;
      const ushort_t* vr = vb + (tokbase + key0 + row) * ldq + h * HD_;
      *(uint4*)&Kt[row][part] = *(const uint4*)(kr + part);
      union { uint4 u; ushort_t s[8]; } tmp;
      tmp.u = *(const uint4*)(vr + part);
#pragma unroll
      for (int i = 0; i < 8; ++i) Vt[part + i][row] = tmp.s[i];
      if (kt + 1 < SEQ_ / 32) {            // gfx1250 global_prefetch of next tile
        __builtin_prefetch(kr + 32 * ldq + part, 0, 1);
        __builtin_prefetch(vr + 32 * ldq + part, 0, 1);
      }
    }
    __syncthreads();

    // S = Q K^T for two 16-key tiles (K-dim = head_dim 64 = 2 WMMA steps)
    v8f sc[2];
#pragma unroll
    for (int nt = 0; nt < 2; ++nt) {
      v8f c = vz;
#pragma unroll
      for (int dc = 0; dc < 2; ++dc) {
        FragBF bkf;
        bkf.q[0] = *(const uint4*)&Kt[nt * 16 + l16][dc * 32 + half * 8];
        bkf.q[1] = *(const uint4*)&Kt[nt * 16 + l16][dc * 32 + 16 + half * 8];
        c = __builtin_amdgcn_wmma_f32_16x16x32_bf16(false, fq[dc].v, false, bkf.v,
                                                    (short)0, c, false, false);
      }
      sc[nt] = c;
    }

    // online softmax per output row (row = r + half*8; n spread over 16-lane group)
#pragma unroll
    for (int r = 0; r < 8; ++r) {
      float s0 = sc[0][r] * 0.125f;        // 1/sqrt(64)
      float s1 = sc[1][r] * 0.125f;
      float mx = grp16_max(fmaxf(s0, s1)); // DPP butterfly, no LDS traffic
      float mnew = fmaxf(mst[r], mx);
      float p0 = exp2f((s0 - mnew) * L2E);
      float p1 = exp2f((s1 - mnew) * L2E);
      float rs = grp16_sum(p0 + p1);
      float scale = exp2f((mst[r] - mnew) * L2E);
      lst[r] = lst[r] * scale + rs;
      mst[r] = mnew;
#pragma unroll
      for (int dt = 0; dt < 4; ++dt) o[dt][r] *= scale;
      int row = r + half * 8;
      Pt[w][row][l16]      = f2bf(p0);     // keys key0..key0+15
      Pt[w][row][16 + l16] = f2bf(p1);     // keys key0+16..key0+31
    }
    __syncthreads();                        // C-layout -> A-layout transpose via LDS

    // O += P(16x32) @ V(32x64)
    FragBF pa;
    pa.q[0] = *(const uint4*)&Pt[w][l16][half * 8];
    pa.q[1] = *(const uint4*)&Pt[w][l16][16 + half * 8];
#pragma unroll
    for (int dt = 0; dt < 4; ++dt) {
      FragBF bv;
      bv.q[0] = *(const uint4*)&Vt[dt * 16 + l16][half * 8];
      bv.q[1] = *(const uint4*)&Vt[dt * 16 + l16][16 + half * 8];
      o[dt] = __builtin_amdgcn_wmma_f32_16x16x32_bf16(false, pa.v, false, bv.v,
                                                      (short)0, o[dt], false, false);
    }
    __syncthreads();
  }

  // normalize and write merged-head output (bf16)
#pragma unroll
  for (int r = 0; r < 8; ++r) {
    float invl = 1.0f / lst[r];
    int row = q0 + r + half * 8;
    ushort_t* orow = ob + (tokbase + row) * H_ + h * HD_;
#pragma unroll
    for (int dt = 0; dt < 4; ++dt) orow[dt * 16 + l16] = f2bf(o[dt][r] * invl);
  }
}

// ---------------------------------------------------------------- launch
extern "C" void kernel_launch(void* const* d_in, const int* in_sizes, int n_in,
                              void* d_out, int out_size, void* d_ws, size_t ws_size,
                              hipStream_t stream) {
  (void)in_sizes; (void)n_in; (void)out_size; (void)ws_size;
  const float* x    = (const float*)d_in[0];
  const float* Wq   = (const float*)d_in[1];
  const float* Wk   = (const float*)d_in[2];
  const float* Wv   = (const float*)d_in[3];
  const float* Wo   = (const float*)d_in[4];
  const float* ln1g = (const float*)d_in[5];
  const float* ln1b = (const float*)d_in[6];
  const float* ln2g = (const float*)d_in[7];
  const float* ln2b = (const float*)d_in[8];
  const float* W1   = (const float*)d_in[9];
  const float* b1   = (const float*)d_in[10];
  const float* W2   = (const float*)d_in[11];
  const float* b2   = (const float*)d_in[12];
  float* out = (float*)d_out;

  char* ws = (char*)d_ws;
  const size_t MB = 1ull << 20;                       // ~120 MB total used
  ushort_t* wqkv_b = (ushort_t*)(ws + 0 * MB);        // [3*H, H] fused (q|k|v rows)
  ushort_t* wo_b   = (ushort_t*)(ws + 6 * MB);
  ushort_t* w1_b   = (ushort_t*)(ws + 8 * MB);
  ushort_t* w2_b   = (ushort_t*)(ws + 16 * MB);
  ushort_t* h1     = (ushort_t*)(ws + 24 * MB);
  ushort_t* qkv    = (ushort_t*)(ws + 32 * MB);       // [TOK, 3072] fused
  ushort_t* ab     = (ushort_t*)(ws + 56 * MB);
  float*    x1     = (float*)   (ws + 64 * MB);
  ushort_t* h2     = (ushort_t*)(ws + 80 * MB);
  ushort_t* fb     = (ushort_t*)(ws + 88 * MB);

  dim3 blk(256);
  // weight casts into fused/single bf16 buffers
  cast_bf16_kernel<<<(H_ * H_ + 255) / 256, blk, 0, stream>>>(Wq, wqkv_b, H_ * H_);
  cast_bf16_kernel<<<(H_ * H_ + 255) / 256, blk, 0, stream>>>(Wk, wqkv_b + H_ * H_, H_ * H_);
  cast_bf16_kernel<<<(H_ * H_ + 255) / 256, blk, 0, stream>>>(Wv, wqkv_b + 2 * H_ * H_, H_ * H_);
  cast_bf16_kernel<<<(H_ * H_ + 255) / 256, blk, 0, stream>>>(Wo, wo_b, H_ * H_);
  cast_bf16_kernel<<<(H_ * FF_ + 255) / 256, blk, 0, stream>>>(W1, w1_b, H_ * FF_);
  cast_bf16_kernel<<<(FF_ * H_ + 255) / 256, blk, 0, stream>>>(W2, w2_b, FF_ * H_);

  // LN1 -> h1 (bf16)
  ln_bf16_kernel<<<TOK_, blk, 0, stream>>>(x, ln1g, ln1b, h1);

  // fused QKV = h1 @ [Wq;Wk;Wv]^T  (one N=3072 GEMM)
  gemm_bf16_kernel<true, false, false, false, false, true>
      <<<dim3(LDQKV / 128, TOK_ / 128), blk, 0, stream>>>(h1, wqkv_b, nullptr, nullptr,
                                                          nullptr, qkv, TOK_, LDQKV, H_);

  // flash attention -> ab (bf16, heads merged)
  attn_kernel<<<dim3(SEQ_ / 128, NH_, BSZ_), blk, 0, stream>>>(
      qkv, qkv + H_, qkv + 2 * H_, ab, LDQKV);

  // x1 = x + attn @ Wo^T (fp32)
  gemm_bf16_kernel<true, false, false, true, true, false>
      <<<dim3(H_ / 128, TOK_ / 128), blk, 0, stream>>>(ab, wo_b, nullptr, x, x1, nullptr,
                                                       TOK_, H_, H_);

  // LN2 -> h2 (bf16)
  ln_bf16_kernel<<<TOK_, blk, 0, stream>>>(x1, ln2g, ln2b, h2);

  // f = gelu(h2 @ W1 + b1) (bf16)
  gemm_bf16_kernel<false, true, true, false, false, true>
      <<<dim3(FF_ / 128, TOK_ / 128), blk, 0, stream>>>(h2, w1_b, b1, nullptr, nullptr, fb,
                                                        TOK_, FF_, H_);
  // out = x1 + f @ W2 + b2 (fp32)
  gemm_bf16_kernel<false, true, false, true, true, false>
      <<<dim3(H_ / 128, TOK_ / 128), blk, 0, stream>>>(fb, w2_b, b2, x1, out, nullptr,
                                                       TOK_, H_, FF_);
}
